// MultiHeadSelfAttention_42417097016029
// MI455X (gfx1250) — matrix operand order repaired and checked by reading, not compiled, not run on hardware
//
#include <hip/hip_runtime.h>
#include <hip/hip_bf16.h>

// ---------------------------------------------------------------------------
// MultiHeadSelfAttention for MI455X (gfx1250).
// B=2, S=2048, D=1024, DV=1024, H=16 (dh=dvh=64).
//
// Pipeline:
//   1) cvt_bf16:    x f32 -> bf16                  (coalesced, 16B in/out)
//   2) transpose_w: W f32 -> W^T bf16              (LDS-tiled)
//   3) proj_gemm:   QK = x@Wqk+b (Q pre-scaled), V = x@Wv+b (head-transposed)
//                   128x128x32 tiles, double-buffered LDS, async global->LDS
//   4) attn_kernel: causal flash attention, 64-row q tiles, bf16 WMMA,
//                   double-buffered async K/V staging, online softmax in LDS
// ---------------------------------------------------------------------------

typedef __attribute__((ext_vector_type(16))) __bf16 v16bf;
typedef __attribute__((ext_vector_type(8)))  __bf16 v8bf;
typedef __attribute__((ext_vector_type(8)))  float  v8f;

#define S_LEN   2048
#define D_DIM   1024
#define DV_DIM  1024
#define H_NUM   16
#define M_ROWS  4096          // B*S
#define NEG_BIG (-1.0e9f)

// ---------------- CDNA5 async global->LDS copy (ASYNCcnt-tracked) ----------
// GLOBAL_LOAD_ASYNC_TO_LDS_B128 (ISA 15.18.3 op 98), GV mode:
//   vdst = per-lane LDS byte offset, vaddr = 64-bit global address.
__device__ __forceinline__ void async_cp16(const void* gp, void* lp) {
  asm volatile("global_load_async_to_lds_b128 %0, %1, off"
               :: "v"((unsigned int)(unsigned long long)lp),
                  "v"((unsigned long long)gp)
               : "memory");
}

__device__ __forceinline__ void wait_async0() {
#if __has_builtin(__builtin_amdgcn_s_wait_asynccnt)
  __builtin_amdgcn_s_wait_asynccnt(0);
#else
  asm volatile("s_wait_asynccnt 0" ::: "memory");
#endif
}

// ---------------- WMMA helpers ---------------------------------------------
__device__ __forceinline__ v8f wmma_bf16(v16bf a, v16bf b, v8f c) {
  return __builtin_amdgcn_wmma_f32_16x16x32_bf16(
      false, a, false, b, (short)0, c, false, false);
}

// A fragment (16x32 bf16) from row-major LDS tile (tile[m][k], stride ld).
// Lanes 0-15: row m, K 0..7 + 16..23; lanes 16-31: row m, K 8..15 + 24..31.
__device__ __forceinline__ v16bf load_a_frag(const __bf16* tile, int ld,
                                             int koff, int lane) {
  const int m  = lane & 15;
  const int lh = lane >> 4;
  const __bf16* p = tile + m * ld + koff + lh * 8;
  v8bf lo = *(const v8bf*)p;
  v8bf hi = *(const v8bf*)(p + 16);
  v16bf a;
#pragma unroll
  for (int i = 0; i < 8; ++i) { a[i] = lo[i]; a[i + 8] = hi[i]; }
  return a;
}

// B fragment (32x16 bf16) from LDS tile stored [n][k] (stride ld).
// Lane n holds column n; lanes 0-15 K 0..7 + 16..23; lanes 16-31 K 8..15 + 24..31.
__device__ __forceinline__ v16bf load_b_frag(const __bf16* tileT, int ld,
                                             int koff, int lane) {
  const int n  = lane & 15;
  const int lh = lane >> 4;
  const __bf16* p = tileT + n * ld + koff + lh * 8;
  v8bf lo = *(const v8bf*)p;
  v8bf hi = *(const v8bf*)(p + 16);
  v16bf r;
#pragma unroll
  for (int i = 0; i < 8; ++i) { r[i] = lo[i]; r[i + 8] = hi[i]; }
  return r;
}

// ---------------------------------------------------------------------------
// Kernel: elementwise f32 -> bf16 (8 elements / thread).
// ---------------------------------------------------------------------------
__global__ __launch_bounds__(256)
void cvt_bf16(const float* __restrict__ in, __bf16* __restrict__ out) {
  const size_t idx = ((size_t)blockIdx.x * 256 + threadIdx.x) * 8;
  float4 a = *(const float4*)&in[idx];
  float4 b = *(const float4*)&in[idx + 4];
  v8bf o = { (__bf16)a.x, (__bf16)a.y, (__bf16)a.z, (__bf16)a.w,
             (__bf16)b.x, (__bf16)b.y, (__bf16)b.z, (__bf16)b.w };
  *(v8bf*)&out[idx] = o;
}

// ---------------------------------------------------------------------------
// Kernel: W f32 [1024 x Ncols] -> W^T bf16 [Ncols x 1024], 64x64 LDS tiles.
// ---------------------------------------------------------------------------
__global__ __launch_bounds__(256)
void transpose_w(const float* __restrict__ W, __bf16* __restrict__ Wt,
                 int Ncols) {
  __shared__ __align__(16) __bf16 T[64][80];   // [n][k], padded rows (160B)
  const int tid = threadIdx.x;
  const int n0 = blockIdx.x * 64;
  const int k0 = blockIdx.y * 64;
#pragma unroll
  for (int i = 0; i < 4; ++i) {
    int slot = tid + i * 256;          // 0..1023
    int r = slot >> 4;                 // k row
    int c = (slot & 15) * 4;           // n col
    float4 v = *(const float4*)&W[(size_t)(k0 + r) * Ncols + n0 + c];
    T[c + 0][r] = (__bf16)v.x;
    T[c + 1][r] = (__bf16)v.y;
    T[c + 2][r] = (__bf16)v.z;
    T[c + 3][r] = (__bf16)v.w;
  }
  __syncthreads();
#pragma unroll
  for (int i = 0; i < 2; ++i) {
    int slot = tid + i * 256;          // 0..511
    int r = slot >> 3;                 // n row
    int c = (slot & 7) * 8;            // k chunk
    *(v8bf*)&Wt[(size_t)(n0 + r) * D_DIM + k0 + c] = *(const v8bf*)&T[r][c];
  }
}

// ---------------------------------------------------------------------------
// Kernel: bf16 GEMM  Y = Xb[M x 1024] @ Wt^T + bias.
// Block 256 threads (8 waves, 4x2). Tile 128(M) x 128(N) x 32(K),
// double-buffered LDS, async global->LDS staging.
// Wave: 32 rows x 64 cols = 2 A-frags x 4 B-frags -> 8 WMMA / k-step.
// vmode 0: row-major bf16 out (QK; cols < scale_limit scaled by qs).
// vmode 1: head-transposed V out: Y[(b*1024 + col)*2048 + s].
// ---------------------------------------------------------------------------
__global__ __launch_bounds__(256)
void proj_gemm(const __bf16* __restrict__ Xb,    // [4096][1024]
               const __bf16* __restrict__ WtT,   // [Ncols][1024]
               const float* __restrict__ bias,   // [Ncols]
               __bf16* __restrict__ Y,
               int Ncols, int scale_limit, float qs, int vmode) {
  __shared__ __align__(32) __bf16 Ab[2][128][32];
  __shared__ __align__(32) __bf16 Bb[2][128][32];

  const int tid  = threadIdx.x;
  const int lane = tid & 31;
  const int wid  = tid >> 5;
  const int row0 = blockIdx.y * 128;
  const int col0 = blockIdx.x * 128;
  const int wr   = (wid >> 1) * 32;   // wave row region
  const int wc   = (wid & 1) * 64;    // wave col region

  v8f acc[2][4] = {};

  auto stage = [&](int buf, int k0) {
#pragma unroll
    for (int i = 0; i < 2; ++i) {
      int slot = tid + i * 256;        // 0..511
      int r = slot >> 2;               // 0..127
      int c = (slot & 3) * 8;          // 16B chunk within 64B row
      async_cp16(&Xb[(size_t)(row0 + r) * D_DIM + k0 + c], &Ab[buf][r][c]);
      async_cp16(&WtT[(size_t)(col0 + r) * D_DIM + k0 + c], &Bb[buf][r][c]);
    }
  };

  stage(0, 0);
  wait_async0();
  __syncthreads();

  int buf = 0;
  for (int k0 = 0; k0 < D_DIM; k0 += 32) {
    if (k0 + 32 < D_DIM) stage(buf ^ 1, k0 + 32);   // overlap with compute

    v16bf a0 = load_a_frag(&Ab[buf][wr][0],      32, 0, lane);
    v16bf a1 = load_a_frag(&Ab[buf][wr + 16][0], 32, 0, lane);
#pragma unroll
    for (int nf = 0; nf < 4; ++nf) {
      v16bf b = load_b_frag(&Bb[buf][wc + nf * 16][0], 32, 0, lane);
      acc[0][nf] = wmma_bf16(a0, b, acc[0][nf]);
      acc[1][nf] = wmma_bf16(a1, b, acc[1][nf]);
    }
    wait_async0();
    __syncthreads();
    buf ^= 1;
  }

  const int lh = lane >> 4;
  const int nl = lane & 15;
#pragma unroll
  for (int ar = 0; ar < 2; ++ar) {
#pragma unroll
    for (int nf = 0; nf < 4; ++nf) {
      int col = col0 + wc + nf * 16 + nl;
      float bc = bias[col];
      if (vmode == 0) {
        float sc = (col < scale_limit) ? qs : 1.0f;
#pragma unroll
        for (int r = 0; r < 8; ++r) {
          int row = row0 + wr + ar * 16 + lh * 8 + r;
          Y[(size_t)row * Ncols + col] = (__bf16)((acc[ar][nf][r] + bc) * sc);
        }
      } else {
        int rowb = row0 + wr + ar * 16 + lh * 8;   // 8 consecutive s values
        int b = rowb >> 11;
        int s0 = rowb & (S_LEN - 1);
        v8bf pk;
#pragma unroll
        for (int r = 0; r < 8; ++r) pk[r] = (__bf16)(acc[ar][nf][r] + bc);
        *(v8bf*)&Y[((size_t)b * DV_DIM + col) * S_LEN + s0] = pk;
      }
    }
  }
}

// ---------------------------------------------------------------------------
// Kernel: causal flash attention, one block per (b, h, 64-row q tile).
// 8 waves (4x2): wave region 16 q-rows x 32 cols. Double-buffered async
// K/V tiles; scores/softmax/P live only in LDS.
// ---------------------------------------------------------------------------
__global__ __launch_bounds__(256)
void attn_kernel(const __bf16* __restrict__ QK,   // [4096][2048]: Q | K
                 const __bf16* __restrict__ Vt,   // [B][1024][2048]
                 float* __restrict__ Out) {       // [4096][1024] f32
  __shared__ __align__(32) __bf16 Qs[64][64];
  __shared__ __align__(32) __bf16 Ks[2][64][64];  // [key][dh]
  __shared__ __align__(32) __bf16 Vs[2][64][64];  // [dv][key]
  __shared__ __align__(32) __bf16 Ps[64][64];
  __shared__ float Sc[64][65];                    // padded rows
  __shared__ float mrow[64], lrow[64], arow[64];

  const int tid  = threadIdx.x;
  const int lane = tid & 31;
  const int wid  = tid >> 5;
  const int lh   = lane >> 4;
  const int nl   = lane & 15;

  const int qt = blockIdx.x;
  const int h  = blockIdx.y;
  const int b  = blockIdx.z;
  const int qb = qt * 64;
  const size_t bS = (size_t)b * S_LEN;

  const int wr = (wid >> 1) * 16;
  const int wc = (wid & 1) * 32;

  auto stageKV = [&](int buf, int kb) {
#pragma unroll
    for (int i = 0; i < 2; ++i) {
      int slot = tid + i * 256;      // 0..511
      int r = slot >> 3;
      int c = (slot & 7) * 8;
      async_cp16(&QK[(bS + kb + r) * (2 * D_DIM) + D_DIM + h * 64 + c],
                 &Ks[buf][r][c]);
      async_cp16(&Vt[((size_t)b * DV_DIM + h * 64 + r) * S_LEN + kb + c],
                 &Vs[buf][r][c]);
    }
  };

  // Stage Q (pre-scaled by 1/sqrt(dvh) in projection) + first K/V tile.
#pragma unroll
  for (int i = 0; i < 2; ++i) {
    int slot = tid + i * 256;
    int r = slot >> 3;
    int c = (slot & 7) * 8;
    async_cp16(&QK[(bS + qb + r) * (2 * D_DIM) + h * 64 + c], &Qs[r][c]);
  }
  stageKV(0, 0);
  if (tid < 64) { mrow[tid] = -3.0e38f; lrow[tid] = 0.0f; }
  wait_async0();
  __syncthreads();

  v8f oacc[2] = {};
  int cur = 0;
  const int nkt = qt + 1;           // causal: key tiles 0..qt

  for (int kt = 0; kt < nkt; ++kt) {
    const int kb = kt * 64;
    if (kt + 1 < nkt) stageKV(cur ^ 1, kb + 64);   // prefetch next tile

    // Scores = Qs @ Ks^T (dh=64 -> two K=32 chunks).
    v8f sacc[2] = {};
#pragma unroll
    for (int kc = 0; kc < 2; ++kc) {
      v16bf a = load_a_frag(&Qs[wr][0], 64, kc * 32, lane);
#pragma unroll
      for (int nf = 0; nf < 2; ++nf) {
        v16bf bm = load_b_frag(&Ks[cur][wc + nf * 16][0], 64, kc * 32, lane);
        sacc[nf] = wmma_bf16(a, bm, sacc[nf]);
      }
    }
#pragma unroll
    for (int nf = 0; nf < 2; ++nf)
#pragma unroll
      for (int r = 0; r < 8; ++r)
        Sc[wr + lh * 8 + r][wc + nf * 16 + nl] = sacc[nf][r];
    __syncthreads();

    // Online softmax: one thread per q row.
    if (tid < 64) {
      const int r = tid;
      const int qrow = qb + r;
      float mold = mrow[r];
      float tmax = -3.0e38f;
      for (int c = 0; c < 64; ++c) {
        float s = (kb + c <= qrow) ? Sc[r][c] : NEG_BIG;
        tmax = fmaxf(tmax, s);
      }
      float mnew = fmaxf(mold, tmax);
      float al = __expf(mold - mnew);
      float ps = 0.0f;
      for (int c = 0; c < 64; ++c) {
        float s = (kb + c <= qrow) ? Sc[r][c] : NEG_BIG;
        float p = __expf(s - mnew);
        Ps[r][c] = (__bf16)p;
        ps += p;
      }
      mrow[r] = mnew;
      lrow[r] = lrow[r] * al + ps;
      arow[r] = al;
    }
    __syncthreads();

    // Rescale running O, then O += P @ V.
#pragma unroll
    for (int r = 0; r < 8; ++r) {
      float al = arow[wr + lh * 8 + r];
      oacc[0][r] *= al;
      oacc[1][r] *= al;
    }
#pragma unroll
    for (int kc = 0; kc < 2; ++kc) {
      v16bf a = load_a_frag(&Ps[wr][0], 64, kc * 32, lane);
#pragma unroll
      for (int nf = 0; nf < 2; ++nf) {
        v16bf bm = load_b_frag(&Vs[cur][wc + nf * 16][0], 64, kc * 32, lane);
        oacc[nf] = wmma_bf16(a, bm, oacc[nf]);
      }
    }
    wait_async0();        // next tile landed; also fences before buffer swap
    __syncthreads();
    cur ^= 1;
  }

  // Epilogue: divide by running sum, write f32 out [b, q, h*64 + dv].
#pragma unroll
  for (int nf = 0; nf < 2; ++nf) {
#pragma unroll
    for (int r = 0; r < 8; ++r) {
      int rl = wr + lh * 8 + r;
      float linv = 1.0f / lrow[rl];
      int qrow = qb + rl;
      int col = h * 64 + wc + nf * 16 + nl;
      Out[(bS + qrow) * DV_DIM + col] = oacc[nf][r] * linv;
    }
  }
}

// ---------------------------------------------------------------------------
extern "C" void kernel_launch(void* const* d_in, const int* in_sizes, int n_in,
                              void* d_out, int out_size, void* d_ws,
                              size_t ws_size, hipStream_t stream) {
  (void)in_sizes; (void)n_in; (void)out_size; (void)ws_size;
  const float* x   = (const float*)d_in[0];
  const float* Wqk = (const float*)d_in[1];
  const float* bqk = (const float*)d_in[2];
  const float* Wv  = (const float*)d_in[3];
  const float* bv  = (const float*)d_in[4];
  float* out = (float*)d_out;

  // Workspace layout (bf16 elements):
  //   Xb   [4096 x 1024]           8 MB
  //   Wqkt [2048 x 1024]           4 MB
  //   Wvt  [1024 x 1024]           2 MB
  //   QK   [4096 x 2048]          16 MB
  //   Vt   [2 x 1024 x 2048]       8 MB
  __bf16* xb   = (__bf16*)d_ws;
  __bf16* wqkt = xb   + (size_t)M_ROWS * D_DIM;
  __bf16* wvt  = wqkt + (size_t)(2 * D_DIM) * D_DIM;
  __bf16* qk   = wvt  + (size_t)DV_DIM * D_DIM;
  __bf16* vt   = qk   + (size_t)M_ROWS * (2 * D_DIM);

  const float qscale = 0.125f;   // 1/sqrt(dvh=64)
  dim3 blk(256);

  cvt_bf16<<<dim3((M_ROWS * D_DIM) / 2048), blk, 0, stream>>>(x, xb);
  transpose_w<<<dim3((2 * D_DIM) / 64, D_DIM / 64), blk, 0, stream>>>(
      Wqk, wqkt, 2 * D_DIM);
  transpose_w<<<dim3(DV_DIM / 64, D_DIM / 64), blk, 0, stream>>>(
      Wv, wvt, DV_DIM);

  // QK projection (Q columns pre-scaled), row-major bf16 out.
  proj_gemm<<<dim3((2 * D_DIM) / 128, M_ROWS / 128), blk, 0, stream>>>(
      xb, wqkt, bqk, qk, 2 * D_DIM, D_DIM, qscale, 0);
  // V projection, head-transposed out.
  proj_gemm<<<dim3(DV_DIM / 128, M_ROWS / 128), blk, 0, stream>>>(
      xb, wvt, bv, vt, DV_DIM, 0, 1.0f, 1);

  attn_kernel<<<dim3(S_LEN / 64, H_NUM, 2), blk, 0, stream>>>(qk, vt, out);
}
